// PhyloEGNN_4166118277824
// MI455X (gfx1250) — compile-verified, matrix operand order checked by reading, not run on hardware
//
#include <hip/hip_runtime.h>

// ---------------------------------------------------------------------------
// PhyloEGNN for MI455X (gfx1250, wave32, WMMA).
// Strategy: fuse all edge MLPs into 16-edge-tile kernels using
// v_wmma_f32_16x16x32_f16 (f32 accumulate), keep activations in LDS, reduce
// messages with global f32 atomics. Node features (5.1MB) are L2-resident, so
// gathers are cheap; we never materialize E x 2H edge tensors (655MB/layer).
// Weights are converted once per launch to f16 transposed [Nout][Kpad] so the
// WMMA B operand is two contiguous 16B loads per lane.
// ---------------------------------------------------------------------------

#define DN   10000
#define DE   640000
#define DH   128
#define DIN  9
#define DOUT 256
#define DB   16
#define KPAD 288   // 2H+3 = 259 padded to 9 K-chunks of 32

typedef __attribute__((ext_vector_type(16))) _Float16 v16h;
typedef __attribute__((ext_vector_type(8)))  _Float16 v8h;
typedef __attribute__((ext_vector_type(8)))  float    v8f;

// ------------------------------- input indices -----------------------------
// setup_inputs() insertion order: x, pos, edge_index, batch, params.
// params leaves flattened jax-style (recursively sorted dict keys).
enum {
  I_X = 0, I_POS, I_EIDX, I_BATCH,
  C1_RELW = 4, C1_RELB, C1_ROOT,          // conv1: rel.W, rel.b, rootW
  C3_RELW, C3_RELB, C3_ROOT,              // conv3
  EG0_BASE = 10,                          // egcl0: 27 leaves (see below)
  EG2_BASE = 37,                          // egcl2
  GT_L1W = 64, GT_L1B, GT_L2W, GT_L2B, GT_L3W, GT_L3B, GT_LNB, GT_LNG,
  OU_L1W = 72, OU_L1B, OU_L2W, OU_L2B, OU_LNB, OU_LNG,
  PR_LINW = 78, PR_LINB, PR_LNB, PR_LNG
};
// egcl leaf offsets (sorted keys: c1,c2,c3,e1,e2,eln1,eln2,ln,n1,n2,nln,scale,w1,w2)
enum {
  E_C1W = 0, E_C1B, E_C2W, E_C2B, E_C3W, E_C3B,
  E_E1W, E_E1B, E_E2W, E_E2B,
  E_ELN1B, E_ELN1G, E_ELN2B, E_ELN2G, E_LNB, E_LNG,
  E_N1W, E_N1B, E_N2W, E_N2B, E_NLNB, E_NLNG,
  E_SCALE, E_W1W, E_W1B, E_W2W, E_W2B
};

// ------------------------------- device helpers ----------------------------
__device__ __forceinline__ float silu_f(float x) { return x / (1.f + expf(-x)); }
__device__ __forceinline__ float gelu_f(float x) {
  return 0.5f * x * (1.f + erff(x * 0.7071067811865476f));
}
__device__ __forceinline__ unsigned fkey(float f) {
  unsigned u = __float_as_uint(f);
  return (u & 0x80000000u) ? ~u : (u | 0x80000000u);
}
__device__ __forceinline__ float funkey(unsigned k) {
  unsigned u = (k & 0x80000000u) ? (k & 0x7fffffffu) : ~k;
  return __uint_as_float(u);
}

// One wave computes a 16x16 f32 tile: acc += A[16xK] * Wt[n0..n0+15][K]^T.
// A layout (ISA 7.12.2, 16-bit A 16x32): lane L holds row M=L&15; its v16h is
// halfs [kb, kb+8) and [kb+16, kb+24) with kb = (L>=16)*8 — two 16B loads.
// Wt stored as [Nout][Kpad] (i.e. W transposed) so B uses the same pattern.
__device__ __forceinline__ v8f wmma_acc_tile(
    v8f acc, const _Float16* __restrict__ A, int ldA,
    const _Float16* __restrict__ Wt, int ldW, int wkoff, int n0, int K) {
  const int lane = threadIdx.x & 31;
  const int row  = lane & 15;
  const int hi   = lane >> 4;
  const _Float16* ar = A + row * ldA + hi * 8;
  const _Float16* br = Wt + (n0 + row) * ldW + wkoff + hi * 8;
  for (int k0 = 0; k0 < K; k0 += 32) {
    v8h alo = *(const v8h*)(ar + k0);
    v8h ahi = *(const v8h*)(ar + k0 + 16);
    v8h blo = *(const v8h*)(br + k0);
    v8h bhi = *(const v8h*)(br + k0 + 16);
    v16h a = __builtin_shufflevector(alo, ahi, 0,1,2,3,4,5,6,7,
                                     8,9,10,11,12,13,14,15);
    v16h b = __builtin_shufflevector(blo, bhi, 0,1,2,3,4,5,6,7,
                                     8,9,10,11,12,13,14,15);
    acc = __builtin_amdgcn_wmma_f32_16x16x32_f16(
        false, a, false, b, (short)0, acc, false, false);
  }
  return acc;
}

// D layout: VGPR r -> M = r + (lane>=16)*8, N = n0 + (lane&15).
__device__ __forceinline__ void store_tile(float* dst, int ldD, int n0, v8f acc) {
  const int lane = threadIdx.x & 31;
  const int col  = n0 + (lane & 15);
  const int rb   = (lane >> 4) * 8;
#pragma unroll
  for (int r = 0; r < 8; ++r) dst[(rb + r) * ldD + col] = acc[r];
}

// Per-row mean / rstd over a 16 x W f32 LDS tile (threads 0..15, one row each).
__device__ __forceinline__ void row_stats(const float* act, int W, int t,
                                          float* sMu, float* sRs) {
  if (t < 16) {
    float mu = 0.f;
    for (int c = 0; c < W; ++c) mu += act[t * W + c];
    mu /= (float)W;
    float v = 0.f;
    for (int c = 0; c < W; ++c) { float d = act[t * W + c] - mu; v += d * d; }
    v /= (float)W;
    sMu[t] = mu;
    sRs[t] = rsqrtf(v + 1e-5f);
  }
}

// ------------------------------- utility kernels ---------------------------
__global__ void k_zero(float* p, int n) {
  int i = blockIdx.x * 256 + threadIdx.x;
  if (i < n) p[i] = 0.f;
}
__global__ void k_copy(const float* s, float* d, int n) {
  int i = blockIdx.x * 256 + threadIdx.x;
  if (i < n) d[i] = s[i];
}
__global__ void k_f32_to_f16(const float* s, _Float16* d, int n) {
  int i = blockIdx.x * 256 + threadIdx.x;
  if (i < n) d[i] = (_Float16)s[i];
}

// W[k][n] (f32, K x Nout) -> Wt[n][k] (f16, NoutP x Kpad, zero padded).
// swapHalf>0: first 2*swapHalf K-rows are half-swapped (x_row <-> x_col) and
// negTail negates the K tail (rel -> -rel) for the backward coord_mlp pass.
__global__ void k_cvt_wt(const float* __restrict__ W, _Float16* __restrict__ Wt,
                         int K, int Nout, int Kpad, int NoutP,
                         int swapHalf, int negTail) {
  int idx = blockIdx.x * 256 + threadIdx.x;
  if (idx >= Kpad * NoutP) return;
  int n = idx / Kpad, k = idx % Kpad;
  float v = 0.f;
  if (k < K && n < Nout) {
    int sk = k;
    float s = 1.f;
    if (swapHalf > 0) {
      if (k < 2 * swapHalf) sk = (k + swapHalf) % (2 * swapHalf);
      else if (negTail)     s = -1.f;
    }
    v = s * W[sk * Nout + n];
  }
  Wt[n * Kpad + k] = (_Float16)v;
}

// ------------------------------- projection --------------------------------
// x[N,9] @ W[9,128] + b -> LN -> GELU.  One block (128 thr) per node.
__global__ __launch_bounds__(128) void k_proj(
    const float* __restrict__ x, const float* __restrict__ W,
    const float* __restrict__ b, const float* __restrict__ lng,
    const float* __restrict__ lnb, float* __restrict__ xo,
    _Float16* __restrict__ xh) {
  __shared__ float xin[DIN];
  __shared__ float red[DH];
  const int n = blockIdx.x, t = threadIdx.x;
  if (t < DIN) xin[t] = x[n * DIN + t];
  __syncthreads();
  float acc = b[t];
  for (int k = 0; k < DIN; ++k) acc += xin[k] * W[k * DH + t];
  red[t] = acc; __syncthreads();
  for (int s = 64; s > 0; s >>= 1) { if (t < s) red[t] += red[t + s]; __syncthreads(); }
  float mean = red[0] / (float)DH; __syncthreads();
  float d = acc - mean;
  red[t] = d * d; __syncthreads();
  for (int s = 64; s > 0; s >>= 1) { if (t < s) red[t] += red[t + s]; __syncthreads(); }
  float y = d * rsqrtf(red[0] / (float)DH + 1e-5f) * lng[t] + lnb[t];
  float g = gelu_f(y);
  xo[(size_t)n * DH + t] = g;
  xh[(size_t)n * DH + t] = (_Float16)g;
}

// ------------------------------- center_rms --------------------------------
__global__ __launch_bounds__(256) void k_pos_stats(const float* __restrict__ pos,
                                                   float* __restrict__ red) {
  __shared__ float s0[256], s1[256], s2[256];
  __shared__ float mean[3];
  const int t = threadIdx.x;
  float sx = 0.f, sy = 0.f, sz = 0.f;
  for (int i = t; i < DN; i += 256) {
    sx += pos[i * 3 + 0]; sy += pos[i * 3 + 1]; sz += pos[i * 3 + 2];
  }
  s0[t] = sx; s1[t] = sy; s2[t] = sz; __syncthreads();
  for (int st = 128; st > 0; st >>= 1) {
    if (t < st) { s0[t] += s0[t + st]; s1[t] += s1[t + st]; s2[t] += s2[t + st]; }
    __syncthreads();
  }
  if (t == 0) { mean[0] = s0[0] / DN; mean[1] = s1[0] / DN; mean[2] = s2[0] / DN; }
  __syncthreads();
  float ss = 0.f;
  for (int i = t; i < DN; i += 256) {
    float dx = pos[i * 3 + 0] - mean[0];
    float dy = pos[i * 3 + 1] - mean[1];
    float dz = pos[i * 3 + 2] - mean[2];
    ss += dx * dx + dy * dy + dz * dz;
  }
  s0[t] = ss; __syncthreads();
  for (int st = 128; st > 0; st >>= 1) { if (t < st) s0[t] += s0[t + st]; __syncthreads(); }
  if (t == 0) {
    float rms = sqrtf(s0[0] / (float)DN);
    red[0] = mean[0]; red[1] = mean[1]; red[2] = mean[2];
    red[3] = 1.f / fmaxf(rms, 1e-6f);
  }
}
__global__ void k_pos_apply(const float* __restrict__ pin,
                            const float* __restrict__ red, float* __restrict__ pout) {
  int i = blockIdx.x * 256 + threadIdx.x;
  if (i >= DN) return;
  for (int k = 0; k < 3; ++k) pout[i * 3 + k] = (pin[i * 3 + k] - red[k]) * red[3];
}

// ------------------------------- EGCL coord --------------------------------
// One block = 16 edges. coord_mlp fwd+bwd via WMMA; bwd uses the pre-swapped
// c1tb so the same A tile [x_row|x_col|rel] serves both directions.
__global__ __launch_bounds__(256) void k_egcl_coord(
    const int* __restrict__ rowI, const int* __restrict__ colI,
    const _Float16* __restrict__ xh, const float* __restrict__ pos,
    float* __restrict__ posOut,
    const _Float16* __restrict__ c1t, const _Float16* __restrict__ c1tb,
    const _Float16* __restrict__ c2t, const _Float16* __restrict__ c3t,
    const float* __restrict__ c1b, const float* __restrict__ c2b,
    const float* __restrict__ c3b,
    const float* __restrict__ w1W, const float* __restrict__ w1b,
    const float* __restrict__ w2W, const float* __restrict__ w2b,
    const float* __restrict__ scaleP) {
  __shared__ __align__(16) _Float16 sA[16 * KPAD];
  __shared__ __align__(16) _Float16 sF[16 * 256];
  __shared__ float sAct[16 * 256];
  __shared__ float sRel[16 * 3];
  __shared__ float sRaw[2][16 * 3];
  __shared__ int sRow[16], sCol[16];
  const int t = threadIdx.x;
  const int e0 = blockIdx.x * 16;
  if (t < 16) { sRow[t] = rowI[e0 + t]; sCol[t] = colI[e0 + t]; }
  __syncthreads();
  {
    const int e = t >> 4, seg = t & 15;
    const v8h* xr = (const v8h*)(xh + (size_t)sRow[e] * DH);
    const v8h* xc = (const v8h*)(xh + (size_t)sCol[e] * DH);
    ((v8h*)(sA + e * KPAD))[seg]      = xr[seg];
    ((v8h*)(sA + e * KPAD + DH))[seg] = xc[seg];
  }
  if (t < 16) {
    int r = sRow[t], c = sCol[t];
    float rx = pos[r * 3 + 0] - pos[c * 3 + 0];
    float ry = pos[r * 3 + 1] - pos[c * 3 + 1];
    float rz = pos[r * 3 + 2] - pos[c * 3 + 2];
    sRel[t * 3 + 0] = rx; sRel[t * 3 + 1] = ry; sRel[t * 3 + 2] = rz;
    sA[t * KPAD + 2 * DH + 0] = (_Float16)rx;
    sA[t * KPAD + 2 * DH + 1] = (_Float16)ry;
    sA[t * KPAD + 2 * DH + 2] = (_Float16)rz;
    for (int k = 2 * DH + 3; k < KPAD; ++k) sA[t * KPAD + k] = (_Float16)0.f;
  }
  __syncthreads();
  const int w = t >> 5;
  for (int dir = 0; dir < 2; ++dir) {
    const _Float16* Wc1 = dir ? c1tb : c1t;
    for (int n0 = w * 16; n0 < 2 * DH; n0 += 128) {      // c1: 288 -> 256
      v8f acc = {};
      acc = wmma_acc_tile(acc, sA, KPAD, Wc1, KPAD, 0, n0, KPAD);
      store_tile(sAct, 2 * DH, n0, acc);
    }
    __syncthreads();
    for (int r = 0; r < 16; ++r)
      sF[r * 256 + t] = (_Float16)silu_f(sAct[r * 256 + t] + c1b[t]);
    __syncthreads();
    {                                                    // c2: 256 -> 128
      v8f acc = {};
      acc = wmma_acc_tile(acc, sF, 256, c2t, 256, 0, w * 16, 256);
      store_tile(sAct, DH, w * 16, acc);
    }
    __syncthreads();
    if (t < DH)
      for (int r = 0; r < 16; ++r)
        sF[r * DH + t] = (_Float16)silu_f(sAct[r * DH + t] + c2b[t]);
    __syncthreads();
    if (w == 0) {                                        // c3: 128 -> 3 (pad16)
      v8f acc = {};
      acc = wmma_acc_tile(acc, sF, DH, c3t, DH, 0, 0, DH);
      store_tile(sAct, 16, 0, acc);
    }
    __syncthreads();
    if (t < 48) {
      int r = t / 3, c = t % 3;
      sRaw[dir][t] = sAct[r * 16 + c] + c3b[c];
    }
    __syncthreads();
  }
  if (t < 16) {                                          // per-edge epilogue
    float rx = sRel[t * 3], ry = sRel[t * 3 + 1], rz = sRel[t * 3 + 2];
    float elen = sqrtf(rx * rx + ry * ry + rz * rz);
    float s = fminf(fmaxf(scaleP[0], 0.f), 5.f);
    for (int dir = 0; dir < 2; ++dir) {
      float sgn = dir ? -1.f : 1.f;
      float r0 = sgn * rx, r1 = sgn * ry, r2 = sgn * rz;
      float acc2 = w2b[0];                               // edge_w MLP (3->128->1)
      for (int j = 0; j < DH; ++j) {
        float h = r0 * w1W[j] + r1 * w1W[DH + j] + r2 * w1W[2 * DH + j] + w1b[j];
        acc2 += silu_f(h) * w2W[j];
      }
      float ew = 1.f / (1.f + expf(-acc2));
      float vx = tanhf(sRaw[dir][t * 3 + 0]);
      float vy = tanhf(sRaw[dir][t * 3 + 1]);
      float vz = tanhf(sRaw[dir][t * 3 + 2]);
      float nrm = fmaxf(sqrtf(vx * vx + vy * vy + vz * vz), 1e-8f);
      float coef = (s / nrm) * 0.05f * elen * ew;
      int node = dir ? sCol[t] : sRow[t];
      atomicAdd(&posOut[node * 3 + 0], vx * coef);
      atomicAdd(&posOut[node * 3 + 1], vy * coef);
      atomicAdd(&posOut[node * 3 + 2], vz * coef);
    }
  }
}

// ------------------------------- EGCL messages -----------------------------
// m = LN(silu(e2(LN(silu(e1([x_r|x_c|rel2])))))); agg[row] += m (fused).
__global__ __launch_bounds__(256) void k_egcl_msg(
    const int* __restrict__ rowI, const int* __restrict__ colI,
    const _Float16* __restrict__ xh, const float* __restrict__ pos,
    const _Float16* __restrict__ e1t, const _Float16* __restrict__ e2t,
    const float* __restrict__ e1b, const float* __restrict__ e2b,
    const float* __restrict__ eln1g, const float* __restrict__ eln1b,
    const float* __restrict__ eln2g, const float* __restrict__ eln2b,
    float* __restrict__ agg) {
  __shared__ __align__(16) _Float16 sA[16 * KPAD];
  __shared__ __align__(16) _Float16 sF[16 * 256];
  __shared__ float sAct[16 * 256];
  __shared__ float sMu[16], sRs[16];
  __shared__ int sRow[16], sCol[16];
  const int t = threadIdx.x;
  const int e0 = blockIdx.x * 16;
  if (t < 16) { sRow[t] = rowI[e0 + t]; sCol[t] = colI[e0 + t]; }
  __syncthreads();
  {
    const int e = t >> 4, seg = t & 15;
    const v8h* xr = (const v8h*)(xh + (size_t)sRow[e] * DH);
    const v8h* xc = (const v8h*)(xh + (size_t)sCol[e] * DH);
    ((v8h*)(sA + e * KPAD))[seg]      = xr[seg];
    ((v8h*)(sA + e * KPAD + DH))[seg] = xc[seg];
  }
  if (t < 16) {
    int r = sRow[t], c = sCol[t];
    float rx = pos[r * 3 + 0] - pos[c * 3 + 0];
    float ry = pos[r * 3 + 1] - pos[c * 3 + 1];
    float rz = pos[r * 3 + 2] - pos[c * 3 + 2];
    sA[t * KPAD + 2 * DH + 0] = (_Float16)rx;
    sA[t * KPAD + 2 * DH + 1] = (_Float16)ry;
    sA[t * KPAD + 2 * DH + 2] = (_Float16)rz;
    for (int k = 2 * DH + 3; k < KPAD; ++k) sA[t * KPAD + k] = (_Float16)0.f;
  }
  __syncthreads();
  const int w = t >> 5;
  for (int n0 = w * 16; n0 < 256; n0 += 128) {           // e1: 288 -> 256
    v8f acc = {};
    acc = wmma_acc_tile(acc, sA, KPAD, e1t, KPAD, 0, n0, KPAD);
    store_tile(sAct, 256, n0, acc);
  }
  __syncthreads();
  for (int r = 0; r < 16; ++r)
    sAct[r * 256 + t] = silu_f(sAct[r * 256 + t] + e1b[t]);
  __syncthreads();
  row_stats(sAct, 256, t, sMu, sRs);
  __syncthreads();
  for (int r = 0; r < 16; ++r)
    sF[r * 256 + t] =
        (_Float16)((sAct[r * 256 + t] - sMu[r]) * sRs[r] * eln1g[t] + eln1b[t]);
  __syncthreads();
  {                                                      // e2: 256 -> 128
    v8f acc = {};
    acc = wmma_acc_tile(acc, sF, 256, e2t, 256, 0, w * 16, 256);
    store_tile(sAct, DH, w * 16, acc);
  }
  __syncthreads();
  if (t < DH)
    for (int r = 0; r < 16; ++r)
      sAct[r * DH + t] = silu_f(sAct[r * DH + t] + e2b[t]);
  __syncthreads();
  row_stats(sAct, DH, t, sMu, sRs);
  __syncthreads();
  if (t < DH)
    for (int r = 0; r < 16; ++r)
      sAct[r * DH + t] =
          (sAct[r * DH + t] - sMu[r]) * sRs[r] * eln2g[t] + eln2b[t];
  __syncthreads();
  {                                                      // agg[row] += m
    const int e = t >> 4, seg = t & 15;
    float* dst = agg + (size_t)sRow[e] * DH + seg * 8;
    const float* src = sAct + e * DH + seg * 8;
    for (int j = 0; j < 8; ++j) atomicAdd(dst + j, src[j]);
  }
}

// ------------------------------- EGCL node update --------------------------
__global__ __launch_bounds__(256) void k_egcl_node(
    const _Float16* __restrict__ xh, const _Float16* __restrict__ aggh,
    const float* __restrict__ xin,
    const _Float16* __restrict__ n1t, const _Float16* __restrict__ n2t,
    const float* __restrict__ n1b, const float* __restrict__ n2b,
    const float* __restrict__ nlng, const float* __restrict__ nlnb,
    const float* __restrict__ lng, const float* __restrict__ lnb,
    float* __restrict__ xout, _Float16* __restrict__ xhout) {
  __shared__ __align__(16) _Float16 sF[16 * 256];
  __shared__ float sAct[16 * 256];
  __shared__ float sMu[16], sRs[16];
  const int t = threadIdx.x, w = t >> 5;
  const int n0 = blockIdx.x * 16;
  const _Float16* Ax = xh + (size_t)n0 * DH;    // rows contiguous in global
  const _Float16* Ag = aggh + (size_t)n0 * DH;
  for (int c0 = w * 16; c0 < 256; c0 += 128) {  // n1: [x|agg] 256 -> 256
    v8f acc = {};
    acc = wmma_acc_tile(acc, Ax, DH, n1t, 256, 0,  c0, DH);
    acc = wmma_acc_tile(acc, Ag, DH, n1t, 256, DH, c0, DH);
    store_tile(sAct, 256, c0, acc);
  }
  __syncthreads();
  for (int r = 0; r < 16; ++r)
    sAct[r * 256 + t] = silu_f(sAct[r * 256 + t] + n1b[t]);
  __syncthreads();
  row_stats(sAct, 256, t, sMu, sRs);
  __syncthreads();
  for (int r = 0; r < 16; ++r)
    sF[r * 256 + t] =
        (_Float16)((sAct[r * 256 + t] - sMu[r]) * sRs[r] * nlng[t] + nlnb[t]);
  __syncthreads();
  {                                             // n2: 256 -> 128
    v8f acc = {};
    acc = wmma_acc_tile(acc, sF, 256, n2t, 256, 0, w * 16, 256);
    store_tile(sAct, DH, w * 16, acc);
  }
  __syncthreads();
  if (t < DH)
    for (int r = 0; r < 16; ++r)                // residual
      sAct[r * DH + t] += n2b[t] + xin[(size_t)(n0 + r) * DH + t];
  __syncthreads();
  row_stats(sAct, DH, t, sMu, sRs);
  __syncthreads();
  if (t < DH)
    for (int r = 0; r < 16; ++r) {
      float y = (sAct[r * DH + t] - sMu[r]) * sRs[r] * lng[t] + lnb[t];
      xout[(size_t)(n0 + r) * DH + t] = y;
      xhout[(size_t)(n0 + r) * DH + t] = (_Float16)y;
    }
}

// ------------------------------- GraphConv ---------------------------------
__global__ void k_scatter_x(const int* __restrict__ rowI, const int* __restrict__ colI,
                            const float* __restrict__ x, float* __restrict__ agg) {
  int idx = blockIdx.x * 256 + threadIdx.x;     // DE*32 threads, 4 floats each
  int e = idx >> 5, lane = idx & 31;
  if (e >= DE) return;
  int r = rowI[e], c = colI[e];
  const float* src = x + (size_t)c * DH + lane * 4;
  float* dst = agg + (size_t)r * DH + lane * 4;
  for (int j = 0; j < 4; ++j) atomicAdd(dst + j, src[j]);
}

__global__ __launch_bounds__(256) void k_gconv(
    const _Float16* __restrict__ xh, const _Float16* __restrict__ aggh,
    const _Float16* __restrict__ relT, const _Float16* __restrict__ rootT,
    const float* __restrict__ relb,
    float* __restrict__ xout, _Float16* __restrict__ xhout) {
  __shared__ float sAct[16 * DH];
  const int t = threadIdx.x, w = t >> 5;
  const int n0 = blockIdx.x * 16;
  {
    v8f acc = {};
    acc = wmma_acc_tile(acc, aggh + (size_t)n0 * DH, DH, relT,  DH, 0, w * 16, DH);
    acc = wmma_acc_tile(acc, xh   + (size_t)n0 * DH, DH, rootT, DH, 0, w * 16, DH);
    store_tile(sAct, DH, w * 16, acc);
  }
  __syncthreads();
  if (t < DH)
    for (int r = 0; r < 16; ++r) {
      float v = sAct[r * DH + t] + relb[t];
      xout[(size_t)(n0 + r) * DH + t] = v;
      xhout[(size_t)(n0 + r) * DH + t] = (_Float16)v;
    }
}

// ------------------------------- gate / pool -------------------------------
__global__ __launch_bounds__(256) void k_gate(
    const _Float16* __restrict__ xh,
    const _Float16* __restrict__ l1t, const float* __restrict__ l1b,
    const float* __restrict__ lng, const float* __restrict__ lnb,
    const _Float16* __restrict__ l2t, const float* __restrict__ l2b,
    const float* __restrict__ l3W, const float* __restrict__ l3b,
    const int* __restrict__ batch, float* __restrict__ gbuf,
    unsigned* __restrict__ gmaxU) {
  __shared__ __align__(16) _Float16 sF[16 * DH];
  __shared__ float sAct[16 * DH];
  __shared__ float sMu[16], sRs[16];
  const int t = threadIdx.x, w = t >> 5;
  const int n0 = blockIdx.x * 16;
  {
    v8f acc = {};
    acc = wmma_acc_tile(acc, xh + (size_t)n0 * DH, DH, l1t, DH, 0, w * 16, DH);
    store_tile(sAct, DH, w * 16, acc);
  }
  __syncthreads();
  if (t < DH) for (int r = 0; r < 16; ++r) sAct[r * DH + t] += l1b[t];
  __syncthreads();
  row_stats(sAct, DH, t, sMu, sRs);
  __syncthreads();
  if (t < DH)
    for (int r = 0; r < 16; ++r) {
      float y = (sAct[r * DH + t] - sMu[r]) * sRs[r] * lng[t] + lnb[t];
      sF[r * DH + t] = (_Float16)gelu_f(y);
    }
  __syncthreads();
  {
    v8f acc = {};
    acc = wmma_acc_tile(acc, sF, DH, l2t, DH, 0, w * 16, DH);
    store_tile(sAct, DH, w * 16, acc);
  }
  __syncthreads();
  if (t < DH)
    for (int r = 0; r < 16; ++r)
      sAct[r * DH + t] = gelu_f(sAct[r * DH + t] + l2b[t]);
  __syncthreads();
  if (t < 16) {
    float g = l3b[0];
    for (int j = 0; j < DH; ++j) g += sAct[t * DH + j] * l3W[j];
    gbuf[n0 + t] = g;
    atomicMax(&gmaxU[batch[n0 + t]], fkey(g));
  }
}

__global__ void k_expsum(const float* __restrict__ gbuf, const int* __restrict__ batch,
                         const unsigned* __restrict__ gmaxU,
                         float* __restrict__ ebuf, float* __restrict__ den) {
  int i = blockIdx.x * 256 + threadIdx.x;
  if (i >= DN) return;
  float ev = expf(gbuf[i] - funkey(gmaxU[batch[i]]));
  ebuf[i] = ev;
  atomicAdd(&den[batch[i]], ev);
}

__global__ void k_pool(const float* __restrict__ x, const float* __restrict__ ebuf,
                       const float* __restrict__ den, const int* __restrict__ batch,
                       float* __restrict__ pooled) {
  int idx = blockIdx.x * 256 + threadIdx.x;     // DN*32 threads, 4 floats each
  int n = idx >> 5, lane = idx & 31;
  if (n >= DN) return;
  float a = ebuf[n] / den[batch[n]];
  const float* src = x + (size_t)n * DH + lane * 4;
  float* dst = pooled + (size_t)batch[n] * DH + lane * 4;
  for (int j = 0; j < 4; ++j) atomicAdd(dst + j, a * src[j]);
}

// ------------------------------- output MLP --------------------------------
__global__ __launch_bounds__(256) void k_out(
    const float* __restrict__ pooled,
    const _Float16* __restrict__ l1t, const float* __restrict__ l1b,
    const float* __restrict__ lng, const float* __restrict__ lnb,
    const _Float16* __restrict__ l2t, const float* __restrict__ l2b,
    float* __restrict__ out) {
  __shared__ __align__(16) _Float16 sP[DB * DH];
  __shared__ __align__(16) _Float16 sF[DB * DH];
  __shared__ float sAct[DB * DOUT];
  __shared__ float sMu[16], sRs[16];
  const int t = threadIdx.x, w = t >> 5;
  for (int i = t; i < DB * DH; i += 256) sP[i] = (_Float16)pooled[i];
  __syncthreads();
  {
    v8f acc = {};
    acc = wmma_acc_tile(acc, sP, DH, l1t, DH, 0, w * 16, DH);
    store_tile(sAct, DH, w * 16, acc);
  }
  __syncthreads();
  if (t < DH) for (int r = 0; r < 16; ++r) sAct[r * DH + t] += l1b[t];
  __syncthreads();
  row_stats(sAct, DH, t, sMu, sRs);
  __syncthreads();
  if (t < DH)
    for (int r = 0; r < 16; ++r) {
      float y = (sAct[r * DH + t] - sMu[r]) * sRs[r] * lng[t] + lnb[t];
      sF[r * DH + t] = (_Float16)gelu_f(y);
    }
  __syncthreads();
  for (int c0 = w * 16; c0 < DOUT; c0 += 128) {
    v8f acc = {};
    acc = wmma_acc_tile(acc, sF, DH, l2t, DH, 0, c0, DH);
    store_tile(sAct, DOUT, c0, acc);
  }
  __syncthreads();
  for (int r = 0; r < 16; ++r)
    out[r * DOUT + t] = sAct[r * DOUT + t] + l2b[t];
}

// ------------------------------- host --------------------------------------
extern "C" void kernel_launch(void* const* d_in, const int* in_sizes, int n_in,
                              void* d_out, int out_size, void* d_ws, size_t ws_size,
                              hipStream_t stream) {
  (void)in_sizes; (void)n_in; (void)out_size; (void)ws_size;
  auto F = [&](int i) { return (const float*)d_in[i]; };
  const int* eidx  = (const int*)d_in[I_EIDX];
  const int* rowI  = eidx;
  const int* colI  = eidx + DE;
  const int* batch = (const int*)d_in[I_BATCH];

  // ---- workspace carve-up ----
  char* base = (char*)d_ws;
  size_t off = 0;
  auto alloc = [&](size_t bytes) -> void* {
    off = (off + 255) & ~(size_t)255;
    void* p = base + off; off += bytes; return p;
  };
  float*    xA    = (float*)alloc((size_t)DN * DH * 4);
  float*    xB    = (float*)alloc((size_t)DN * DH * 4);
  _Float16* xh    = (_Float16*)alloc((size_t)DN * DH * 2);
  float*    agg   = (float*)alloc((size_t)DN * DH * 4);
  _Float16* aggh  = (_Float16*)alloc((size_t)DN * DH * 2);
  float*    posA  = (float*)alloc((size_t)DN * 3 * 4);
  float*    posB  = (float*)alloc((size_t)DN * 3 * 4);
  float*    red   = (float*)alloc(64 * 4);
  float*    gbuf  = (float*)alloc((size_t)DN * 4);
  float*    ebuf  = (float*)alloc((size_t)DN * 4);
  unsigned* gmaxU = (unsigned*)alloc(DB * 4);
  float*    den   = (float*)alloc(DB * 4);
  float*    pooled= (float*)alloc((size_t)DB * DH * 4);
  _Float16 *c1t[2], *c1tb[2], *c2t[2], *c3t[2], *e1t[2], *e2t[2], *n1t[2], *n2t[2];
  for (int s = 0; s < 2; ++s) {
    c1t[s]  = (_Float16*)alloc((size_t)256 * KPAD * 2);
    c1tb[s] = (_Float16*)alloc((size_t)256 * KPAD * 2);
    c2t[s]  = (_Float16*)alloc((size_t)128 * 256 * 2);
    c3t[s]  = (_Float16*)alloc((size_t)16 * 128 * 2);
    e1t[s]  = (_Float16*)alloc((size_t)256 * KPAD * 2);
    e2t[s]  = (_Float16*)alloc((size_t)128 * 256 * 2);
    n1t[s]  = (_Float16*)alloc((size_t)256 * 256 * 2);
    n2t[s]  = (_Float16*)alloc((size_t)128 * 256 * 2);
  }
  _Float16 *relT[2], *rootT[2];
  for (int s = 0; s < 2; ++s) {
    relT[s]  = (_Float16*)alloc((size_t)128 * 128 * 2);
    rootT[s] = (_Float16*)alloc((size_t)128 * 128 * 2);
  }
  _Float16* gl1t = (_Float16*)alloc((size_t)128 * 128 * 2);
  _Float16* gl2t = (_Float16*)alloc((size_t)128 * 128 * 2);
  _Float16* ol1t = (_Float16*)alloc((size_t)128 * 128 * 2);
  _Float16* ol2t = (_Float16*)alloc((size_t)256 * 128 * 2);

  auto cvt = [&](const float* W, _Float16* Wt, int K, int Nout, int Kp, int NP,
                 int swapH, int neg) {
    int tot = Kp * NP;
    k_cvt_wt<<<dim3((tot + 255) / 256), dim3(256), 0, stream>>>(
        W, Wt, K, Nout, Kp, NP, swapH, neg);
  };
  auto zero = [&](float* p, int n) {
    k_zero<<<dim3((n + 255) / 256), dim3(256), 0, stream>>>(p, n);
  };

  // ---- weight conversion (f32 -> f16 transposed, zero-padded) ----
  const int EB[2] = {EG0_BASE, EG2_BASE};
  for (int s = 0; s < 2; ++s) {
    int b = EB[s];
    cvt(F(b + E_C1W), c1t[s],  2 * DH + 3, 256, KPAD, 256, 0, 0);
    cvt(F(b + E_C1W), c1tb[s], 2 * DH + 3, 256, KPAD, 256, DH, 1); // bwd pass
    cvt(F(b + E_C2W), c2t[s],  256, 128, 256, 128, 0, 0);
    cvt(F(b + E_C3W), c3t[s],  128, 3,   128, 16,  0, 0);
    cvt(F(b + E_E1W), e1t[s],  2 * DH + 3, 256, KPAD, 256, 0, 0);
    cvt(F(b + E_E2W), e2t[s],  256, 128, 256, 128, 0, 0);
    cvt(F(b + E_N1W), n1t[s],  256, 256, 256, 256, 0, 0);
    cvt(F(b + E_N2W), n2t[s],  256, 128, 256, 128, 0, 0);
  }
  cvt(F(C1_RELW), relT[0], 128, 128, 128, 128, 0, 0);
  cvt(F(C1_ROOT), rootT[0], 128, 128, 128, 128, 0, 0);
  cvt(F(C3_RELW), relT[1], 128, 128, 128, 128, 0, 0);
  cvt(F(C3_ROOT), rootT[1], 128, 128, 128, 128, 0, 0);
  cvt(F(GT_L1W), gl1t, 128, 128, 128, 128, 0, 0);
  cvt(F(GT_L2W), gl2t, 128, 128, 128, 128, 0, 0);
  cvt(F(OU_L1W), ol1t, 128, 128, 128, 128, 0, 0);
  cvt(F(OU_L2W), ol2t, 128, 256, 128, 256, 0, 0);

  // ---- projection + initial center_rms ----
  k_proj<<<dim3(DN), dim3(DH), 0, stream>>>(F(I_X), F(PR_LINW), F(PR_LINB),
                                            F(PR_LNG), F(PR_LNB), xA, xh);
  k_pos_stats<<<dim3(1), dim3(256), 0, stream>>>(F(I_POS), red);
  k_pos_apply<<<dim3((DN + 255) / 256), dim3(256), 0, stream>>>(F(I_POS), red, posA);

  float *xcur = xA, *xnext = xB;
  float *pcur = posA, *pother = posB;
  for (int s = 0; s < 2; ++s) {                 // { egcl, center, conv, center }
    int b = EB[s];
    // EGCL: coordinate update (pos' = pos + atomic deltas)
    k_copy<<<dim3((DN * 3 + 255) / 256), dim3(256), 0, stream>>>(pcur, pother, DN * 3);
    k_egcl_coord<<<dim3(DE / 16), dim3(256), 0, stream>>>(
        rowI, colI, xh, pcur, pother, c1t[s], c1tb[s], c2t[s], c3t[s],
        F(b + E_C1B), F(b + E_C2B), F(b + E_C3B),
        F(b + E_W1W), F(b + E_W1B), F(b + E_W2W), F(b + E_W2B), F(b + E_SCALE));
    // EGCL: fused edge messages + segment_sum
    zero(agg, DN * DH);
    k_egcl_msg<<<dim3(DE / 16), dim3(256), 0, stream>>>(
        rowI, colI, xh, pother, e1t[s], e2t[s], F(b + E_E1B), F(b + E_E2B),
        F(b + E_ELN1G), F(b + E_ELN1B), F(b + E_ELN2G), F(b + E_ELN2B), agg);
    k_f32_to_f16<<<dim3((DN * DH + 255) / 256), dim3(256), 0, stream>>>(agg, aggh, DN * DH);
    // EGCL: node update (residual + LN)
    k_egcl_node<<<dim3(DN / 16), dim3(256), 0, stream>>>(
        xh, aggh, xcur, n1t[s], n2t[s], F(b + E_N1B), F(b + E_N2B),
        F(b + E_NLNG), F(b + E_NLNB), F(b + E_LNG), F(b + E_LNB), xnext, xh);
    { float* tmp = xcur; xcur = xnext; xnext = tmp; }
    // center_rms after EGCL
    k_pos_stats<<<dim3(1), dim3(256), 0, stream>>>(pother, red);
    k_pos_apply<<<dim3((DN + 255) / 256), dim3(256), 0, stream>>>(pother, red, pcur);
    // GraphConv
    zero(agg, DN * DH);
    k_scatter_x<<<dim3(DE * 32 / 256), dim3(256), 0, stream>>>(rowI, colI, xcur, agg);
    k_f32_to_f16<<<dim3((DN * DH + 255) / 256), dim3(256), 0, stream>>>(agg, aggh, DN * DH);
    k_gconv<<<dim3(DN / 16), dim3(256), 0, stream>>>(
        xh, aggh, relT[s], rootT[s], (s == 0) ? F(C1_RELB) : F(C3_RELB), xnext, xh);
    { float* tmp = xcur; xcur = xnext; xnext = tmp; }
    // center_rms after conv (feeds next EGCL)
    k_pos_stats<<<dim3(1), dim3(256), 0, stream>>>(pcur, red);
    k_pos_apply<<<dim3((DN + 255) / 256), dim3(256), 0, stream>>>(pcur, red, pother);
    { float* tmp = pcur; pcur = pother; pother = tmp; }
  }

  // ---- global attention pool ----
  zero((float*)gmaxU, DB);
  zero(den, DB);
  zero(pooled, DB * DH);
  k_gate<<<dim3(DN / 16), dim3(256), 0, stream>>>(
      xh, gl1t, F(GT_L1B), F(GT_LNG), F(GT_LNB), gl2t, F(GT_L2B),
      F(GT_L3W), F(GT_L3B), batch, gbuf, gmaxU);
  k_expsum<<<dim3((DN + 255) / 256), dim3(256), 0, stream>>>(gbuf, batch, gmaxU, ebuf, den);
  k_pool<<<dim3(DN * 32 / 256), dim3(256), 0, stream>>>(xcur, ebuf, den, batch, pooled);

  // ---- output MLP ----
  k_out<<<dim3(1), dim3(256), 0, stream>>>(
      pooled, ol1t, F(OU_L1B), F(OU_LNG), F(OU_LNB), ol2t, F(OU_L2B), (float*)d_out);
}
